// SelfAttention_68341519614380
// MI455X (gfx1250) — compile-verified, matrix operand order checked by reading, not run on hardware
//
#include <hip/hip_runtime.h>

#define EMBED 1024
#define HEADS 16
#define HEAD_DIM 64
#define NBATCH 4
#define SEQ 2048

typedef __attribute__((ext_vector_type(16))) _Float16 v16h;
typedef __attribute__((ext_vector_type(8)))  _Float16 v8h;
typedef __attribute__((ext_vector_type(8)))  float    v8f;

static __device__ __forceinline__ v16h cat16(v8h a, v8h b) {
  return __builtin_shufflevector(a, b, 0,1,2,3,4,5,6,7,8,9,10,11,12,13,14,15);
}
static __device__ __forceinline__ v8h ldh8(const _Float16* p) { return *(const v8h*)p; }
static __device__ __forceinline__ v8h cvt8(const float* p) {
  float4 a = *(const float4*)p;
  float4 b = *(const float4*)(p + 4);
  v8h r;
  r[0]=(_Float16)a.x; r[1]=(_Float16)a.y; r[2]=(_Float16)a.z; r[3]=(_Float16)a.w;
  r[4]=(_Float16)b.x; r[5]=(_Float16)b.y; r[6]=(_Float16)b.z; r[7]=(_Float16)b.w;
  return r;
}
#define WMMA_F16(a,b,c) __builtin_amdgcn_wmma_f32_16x16x32_f16(false,(a),false,(b),(short)0,(c),false,false)

// Async global->LDS copy, 16 bytes per lane (GLOBAL_LOAD_ASYNC_TO_LDS_B128,
// tracked by ASYNCcnt). LDS aperture keeps the byte offset in addr[31:0], so
// truncating the generic pointer yields the LDS-address VGPR value.
static __device__ __forceinline__ void async_cp16(_Float16* ldst, const _Float16* gsrc) {
  unsigned lofs = (unsigned)(unsigned long long)ldst;
  asm volatile("global_load_async_to_lds_b128 %0, %1, off"
               :: "v"(lofs), "v"(gsrc) : "memory");
}
static __device__ __forceinline__ void wait_async_le16() {
  asm volatile("s_wait_asynccnt 16" ::: "memory");
}
static __device__ __forceinline__ void wait_async_0() {
  asm volatile("s_wait_asynccnt 0" ::: "memory");
}

// ---------------------------------------------------------------------------
// Kernel 0: Wo f32 -> f16
// ---------------------------------------------------------------------------
__global__ __launch_bounds__(256) void wo_cvt_kernel(const float* __restrict__ Wo,
                                                     _Float16* __restrict__ Wo16) {
  int i = blockIdx.x * 256 + threadIdx.x;
  if (i < EMBED * EMBED) Wo16[i] = (_Float16)Wo[i];
}

// ---------------------------------------------------------------------------
// Kernel 1: per-head projection  Y[16x64] = X_tile[16x64] @ W^T
//   transposed==0 : store row-major  out[(n*H+h)*S + s][d]      (Q, K)
//   transposed==1 : store transposed out[(n*H+h)*64 + d][s]     (V)
// One wave (32 threads) per (token-tile, head).
// ---------------------------------------------------------------------------
__global__ __launch_bounds__(32) void proj_kernel(const float* __restrict__ X,
                                                  const float* __restrict__ W,
                                                  _Float16* __restrict__ outp,
                                                  int transposed) {
  const int lane = threadIdx.x;
  const int hw   = lane >> 4;          // half-wave: 0 or 1
  const int nl   = lane & 15;
  const int lo   = hw * 8;             // K-chunk base inside A/B fragment

  const int h  = blockIdx.x & (HEADS - 1);
  const int st = blockIdx.x / HEADS;   // token tile within batch
  const int bn = blockIdx.y;
  const int s0 = st * 16;

  // A fragments (tokens x d), K = 64 -> 2 k-steps of 32.
  v16h af[2];
  {
    const float* xr = X + ((size_t)(bn * SEQ + s0 + nl)) * EMBED + h * HEAD_DIM;
#pragma unroll
    for (int ks = 0; ks < 2; ++ks)
      af[ks] = cat16(cvt8(xr + ks * 32 + lo), cvt8(xr + ks * 32 + 16 + lo));
  }

  // 4 N-tiles of 16 output dims each.
  v8f acc[4];
#pragma unroll
  for (int nt = 0; nt < 4; ++nt) {
    v8f c = {};
#pragma unroll
    for (int ks = 0; ks < 2; ++ks) {
      // B[k][n] = W[e0+n][k]  (Y = X @ W^T); contiguous along k.
      const float* wr = W + (size_t)(nt * 16 + nl) * HEAD_DIM + ks * 32 + lo;
      v16h b = cat16(cvt8(wr), cvt8(wr + 16));
      c = WMMA_F16(af[ks], b, c);
    }
    acc[nt] = c;
  }

  if (transposed) {
    // C-frag: lane holds col e=nt*16+nl, rows s0+hw*8+r (r=0..7) -> contiguous s.
#pragma unroll
    for (int nt = 0; nt < 4; ++nt) {
      v8h pk;
#pragma unroll
      for (int r = 0; r < 8; ++r) pk[r] = (_Float16)acc[nt][r];
      _Float16* dst = outp + ((size_t)(bn * HEADS + h) * HEAD_DIM + nt * 16 + nl) * SEQ
                           + s0 + hw * 8;
      *(v8h*)dst = pk;
    }
  } else {
    __shared__ __align__(16) _Float16 tile[16 * 64];
#pragma unroll
    for (int nt = 0; nt < 4; ++nt)
#pragma unroll
      for (int r = 0; r < 8; ++r)
        tile[(hw * 8 + r) * 64 + nt * 16 + nl] = (_Float16)acc[nt][r];
    // same-wave LDS ops are in-order: loads below observe the stores above
    const int row = lane >> 1, c0 = (lane & 1) * 32;
    _Float16* dst = outp + ((size_t)(bn * HEADS + h) * SEQ + s0 + row) * HEAD_DIM + c0;
    const _Float16* src = &tile[row * 64 + c0];
#pragma unroll
    for (int i = 0; i < 4; ++i) *(v8h*)(dst + i * 8) = *(const v8h*)(src + i * 8);
  }
}

// ---------------------------------------------------------------------------
// Kernel 2: flash attention, one wave per (batch, head, 16-query tile).
// Key tiles of 32 keys. K/V tiles are double-buffered in LDS and filled with
// GLOBAL_LOAD_ASYNC_TO_LDS_B128 (ASYNCcnt), overlapping the WMMA + softmax of
// the current tile with the DMA of the next.
// ---------------------------------------------------------------------------
__global__ __launch_bounds__(32) void attn_kernel(const _Float16* __restrict__ Qp,
                                                  const _Float16* __restrict__ Kp,
                                                  const _Float16* __restrict__ Vt,
                                                  const int* __restrict__ mask,
                                                  _Float16* __restrict__ AO) {
  const int lane = threadIdx.x;
  const int hw   = lane >> 4;
  const int nl   = lane & 15;
  const int lo   = hw * 8;

  const int h  = blockIdx.x & (HEADS - 1);
  const int qt = blockIdx.x / HEADS;
  const int bn = blockIdx.y;
  const int q0 = qt * 16;

  const _Float16* Qh = Qp + (size_t)(bn * HEADS + h) * SEQ * HEAD_DIM;
  const _Float16* Kh = Kp + (size_t)(bn * HEADS + h) * SEQ * HEAD_DIM;
  const _Float16* Vh = Vt + (size_t)(bn * HEADS + h) * HEAD_DIM * SEQ;

  // Double-buffered K/V tiles: K row-major [32 keys][64 d], V [64 d][32 keys].
  __shared__ __align__(16) _Float16 Kbuf[2][32 * 64];
  __shared__ __align__(16) _Float16 Vbuf[2][64 * 32];
  __shared__ __align__(16) _Float16 pls[16 * 32];

  // Q A-fragments (row m = nl on both half-waves).
  v16h qa[2];
  {
    const _Float16* qr = Qh + (size_t)(q0 + nl) * HEAD_DIM;
#pragma unroll
    for (int ks = 0; ks < 2; ++ks)
      qa[ks] = cat16(ldh8(qr + ks * 32 + lo), ldh8(qr + ks * 32 + 16 + lo));
  }

  v8f o[4] = {v8f{}, v8f{}, v8f{}, v8f{}};
  float rmax[8], rsum[8];
#pragma unroll
  for (int r = 0; r < 8; ++r) { rmax[r] = -1.0e30f; rsum[r] = 0.0f; }

  const float scale = 0.03125f;              // 1/sqrt(EMBED) = 1/32
  const float mfill = -1.0e20f * 0.03125f;   // masked value after scaling

  const int NT = SEQ / 32;

  // Issue async fill of tile 0: 8 instrs for K (lane = key row, chunk i),
  // 8 instrs for V (lane covers d-rows lane and lane+32). 16 per tile.
  {
    const int k0 = 0;
#pragma unroll
    for (int i = 0; i < 8; ++i)
      async_cp16(&Kbuf[0][lane * 64 + i * 8], Kh + (size_t)(k0 + lane) * HEAD_DIM + i * 8);
#pragma unroll
    for (int j = 0; j < 2; ++j)
#pragma unroll
      for (int i = 0; i < 4; ++i) {
        const int dr = lane + j * 32;
        async_cp16(&Vbuf[0][dr * 32 + i * 8], Vh + (size_t)dr * SEQ + k0 + i * 8);
      }
  }

#pragma unroll 1
  for (int kt = 0; kt < NT; ++kt) {
    const int k0  = kt * 32;
    const int cur = kt & 1;

    // Kick off next tile's DMA, then wait for the current tile (async loads
    // retire in order: <=16 outstanding => previous 16 have landed in LDS).
    if (kt + 1 < NT) {
      const int nxt = (kt + 1) & 1;
      const int kn  = k0 + 32;
#pragma unroll
      for (int i = 0; i < 8; ++i)
        async_cp16(&Kbuf[nxt][lane * 64 + i * 8], Kh + (size_t)(kn + lane) * HEAD_DIM + i * 8);
#pragma unroll
      for (int j = 0; j < 2; ++j)
#pragma unroll
        for (int i = 0; i < 4; ++i) {
          const int dr = lane + j * 32;
          async_cp16(&Vbuf[nxt][dr * 32 + i * 8], Vh + (size_t)dr * SEQ + kn + i * 8);
        }
      wait_async_le16();
    } else {
      wait_async_0();
    }

    // ---- S[16x32] = Q @ K^T : two 16-col tiles (B-frags from LDS) --------
    v8f sf[2];
#pragma unroll
    for (int c = 0; c < 2; ++c) {
      v8f s = {};
      const _Float16* kr = &Kbuf[cur][(c * 16 + nl) * 64];
#pragma unroll
      for (int ks = 0; ks < 2; ++ks) {
        v16h b = cat16(ldh8(kr + ks * 32 + lo), ldh8(kr + ks * 32 + 16 + lo));
        s = WMMA_F16(qa[ks], b, s);
      }
      sf[c] = s;
    }

    // ---- scale + mask ----------------------------------------------------
#pragma unroll
    for (int c = 0; c < 2; ++c)
#pragma unroll
      for (int r = 0; r < 8; ++r) {
        const int row = q0 + hw * 8 + r;
        const int key = k0 + c * 16 + nl;
        const int mv = mask[((size_t)bn * SEQ + row) * SEQ + key];
        sf[c][r] = mv ? sf[c][r] * scale : mfill;
      }

    // ---- online softmax: row max over 32 cols (half-wave reduction) -----
    float nm[8], alpha[8];
#pragma unroll
    for (int r = 0; r < 8; ++r) {
      float m = fmaxf(sf[0][r], sf[1][r]);
      m = fmaxf(m, __shfl_xor(m, 1, 32));
      m = fmaxf(m, __shfl_xor(m, 2, 32));
      m = fmaxf(m, __shfl_xor(m, 4, 32));
      m = fmaxf(m, __shfl_xor(m, 8, 32));
      nm[r] = fmaxf(rmax[r], m);
      alpha[r] = __expf(rmax[r] - nm[r]);
      rmax[r] = nm[r];
    }

    // ---- P = exp(S - max): to LDS (C-layout -> A-layout bounce) ---------
    float part[8];
#pragma unroll
    for (int r = 0; r < 8; ++r) {
      float p0 = __expf(sf[0][r] - nm[r]);
      float p1 = __expf(sf[1][r] - nm[r]);
      part[r] = p0 + p1;
      pls[(hw * 8 + r) * 32 + nl]      = (_Float16)p0;
      pls[(hw * 8 + r) * 32 + 16 + nl] = (_Float16)p1;
    }
#pragma unroll
    for (int r = 0; r < 8; ++r) {
      float t = part[r];
      t += __shfl_xor(t, 1, 32);
      t += __shfl_xor(t, 2, 32);
      t += __shfl_xor(t, 4, 32);
      t += __shfl_xor(t, 8, 32);
      rsum[r] = rsum[r] * alpha[r] + t;
    }

    // ---- rescale accumulator --------------------------------------------
#pragma unroll
    for (int dt = 0; dt < 4; ++dt)
#pragma unroll
      for (int r = 0; r < 8; ++r) o[dt][r] *= alpha[r];

    // ---- O += P @ V (P A-frag and V B-frags from LDS) --------------------
    v16h pa = cat16(ldh8(&pls[nl * 32 + lo]), ldh8(&pls[nl * 32 + 16 + lo]));
#pragma unroll
    for (int dt = 0; dt < 4; ++dt) {
      const _Float16* vr = &Vbuf[cur][(dt * 16 + nl) * 32];
      v16h b = cat16(ldh8(vr + lo), ldh8(vr + 16 + lo));
      o[dt] = WMMA_F16(pa, b, o[dt]);
    }
  }

  // ---- finalize: O / rowsum, bounce to LDS, contiguous f16 store --------
  __shared__ __align__(16) _Float16 ot[16 * 64];
#pragma unroll
  for (int dt = 0; dt < 4; ++dt)
#pragma unroll
    for (int r = 0; r < 8; ++r)
      ot[(hw * 8 + r) * 64 + dt * 16 + nl] = (_Float16)(o[dt][r] * (1.0f / rsum[r]));

  const int row = lane >> 1, c0 = (lane & 1) * 32;
  _Float16* dst = AO + ((size_t)bn * SEQ + q0 + row) * EMBED + h * HEAD_DIM + c0;
  const _Float16* src = &ot[row * 64 + c0];
#pragma unroll
  for (int i = 0; i < 4; ++i) *(v8h*)(dst + i * 8) = *(const v8h*)(src + i * 8);
}

// ---------------------------------------------------------------------------
// Kernel 3: Y[8192x1024] = AO @ Wo^T + bo.  One wave per 16x64 output tile.
// ---------------------------------------------------------------------------
__global__ __launch_bounds__(32) void outproj_kernel(const _Float16* __restrict__ AO,
                                                     const _Float16* __restrict__ Wo16,
                                                     const float* __restrict__ bo,
                                                     float* __restrict__ Y) {
  const int lane = threadIdx.x;
  const int hw   = lane >> 4;
  const int nl   = lane & 15;
  const int lo   = hw * 8;

  const int et = blockIdx.x & 15;      // 16 tiles of 64 output columns
  const int tt = blockIdx.x >> 4;      // N*S/16 = 512 token tiles
  const int t0 = tt * 16;
  const int e0 = et * 64;

  v8f acc[4] = {v8f{}, v8f{}, v8f{}, v8f{}};

#pragma unroll 1
  for (int k = 0; k < EMBED; k += 32) {
    const _Float16* ar = AO + (size_t)(t0 + nl) * EMBED + k;
    v16h a = cat16(ldh8(ar + lo), ldh8(ar + 16 + lo));
#pragma unroll
    for (int nt = 0; nt < 4; ++nt) {
      const _Float16* wr = Wo16 + (size_t)(e0 + nt * 16 + nl) * EMBED + k;
      v16h b = cat16(ldh8(wr + lo), ldh8(wr + 16 + lo));
      acc[nt] = WMMA_F16(a, b, acc[nt]);
    }
  }

  __shared__ __align__(16) float tile[16 * 64];
#pragma unroll
  for (int nt = 0; nt < 4; ++nt) {
    const float bv = bo[e0 + nt * 16 + nl];
#pragma unroll
    for (int r = 0; r < 8; ++r)
      tile[(hw * 8 + r) * 64 + nt * 16 + nl] = acc[nt][r] + bv;
  }
  const int row = lane >> 1, c0 = (lane & 1) * 32;
  float4* dst = (float4*)(Y + (size_t)(t0 + row) * EMBED + e0 + c0);
  const float4* src = (const float4*)&tile[row * 64 + c0];
#pragma unroll
  for (int i = 0; i < 8; ++i) dst[i] = src[i];
}

// ---------------------------------------------------------------------------
extern "C" void kernel_launch(void* const* d_in, const int* in_sizes, int n_in,
                              void* d_out, int out_size, void* d_ws, size_t ws_size,
                              hipStream_t stream) {
  const float* values = (const float*)d_in[0];
  const float* keys_  = (const float*)d_in[1];
  const float* query  = (const float*)d_in[2];
  const int*   mask   = (const int*)d_in[3];
  const float* Wv     = (const float*)d_in[4];
  const float* Wk     = (const float*)d_in[5];
  const float* Wq     = (const float*)d_in[6];
  const float* Wo     = (const float*)d_in[7];
  const float* bo     = (const float*)d_in[8];
  float* out = (float*)d_out;

  // Workspace layout (bytes):
  //   Qp [N][H][S][D]  f16 : 16 MB
  //   Kp [N][H][S][D]  f16 : 16 MB
  //   Vt [N][H][D][S]  f16 : 16 MB   (transposed for P@V B-fragments)
  //   AO [N][S][E]     f16 : 16 MB
  //   Wo16 [E][E]      f16 :  2 MB
  const size_t QKV = (size_t)NBATCH * HEADS * SEQ * HEAD_DIM * sizeof(_Float16);
  char* ws = (char*)d_ws;
  _Float16* Qp   = (_Float16*)(ws);
  _Float16* Kp   = (_Float16*)(ws + QKV);
  _Float16* Vt   = (_Float16*)(ws + 2 * QKV);
  _Float16* AO   = (_Float16*)(ws + 3 * QKV);
  _Float16* Wo16 = (_Float16*)(ws + 4 * QKV);

  wo_cvt_kernel<<<(EMBED * EMBED) / 256, 256, 0, stream>>>(Wo, Wo16);

  dim3 gproj((SEQ / 16) * HEADS, NBATCH);
  proj_kernel<<<gproj, 32, 0, stream>>>(query,  Wq, Qp, 0);
  proj_kernel<<<gproj, 32, 0, stream>>>(keys_,  Wk, Kp, 0);
  proj_kernel<<<gproj, 32, 0, stream>>>(values, Wv, Vt, 1);

  dim3 gattn((SEQ / 16) * HEADS, NBATCH);
  attn_kernel<<<gattn, 32, 0, stream>>>(Qp, Kp, Vt, mask, AO);

  outproj_kernel<<<(NBATCH * SEQ / 16) * (EMBED / 64), 32, 0, stream>>>(AO, Wo16, bo, out);
}
